// k1_GNN_sub_sep_ppgn_7842610283376
// MI455X (gfx1250) — matrix operand order repaired and checked by reading, not compiled, not run on hardware
//
#include <hip/hip_runtime.h>
#include <hip/hip_bf16.h>
#include <math.h>

#define BB 64
#define NNp 64
#define SSp 10
#define NSUBc (BB*NNp)          /* 4096  */
#define NNODEc (NSUBc*SSp)      /* 40960 */
#define EDGc  (NNODEc*8)        /* 327680 */
#define EORIGc 65536

typedef _Float16 h16;
typedef __attribute__((ext_vector_type(16))) _Float16 v16h;
typedef __attribute__((ext_vector_type(8)))  _Float16 h16v8;
typedef __attribute__((ext_vector_type(8)))  float    v8f;

__device__ __forceinline__ v8f wmma16(v16h a, v16h b, v8f c) {
  // D = A(16x32 f16) * B(32x16 f16) + C(16x16 f32)
  return __builtin_amdgcn_wmma_f32_16x16x32_f16(false, a, false, b, (short)0, c,
                                                false, false);
}
// A-fragment element j of lane-half hi -> logical K within the 32-chunk (contiguous order, matching the B packs)
__device__ __forceinline__ int kAidx(int j, int hi) { return j + 16 * hi; }

// ---- CDNA5 async copy: 64B per lane per call (2x32B, 4x b128), LDS<-global.
// IOFFSET applies to both LDS and global address (ISA 08_async_tensor 4.4),
// so a mirrored layout needs only one base address pair.
__device__ __forceinline__ void async_cp32B_x2(unsigned lds_dst, const h16* gsrc) {
  unsigned long long ga = (unsigned long long)gsrc;
  asm volatile("global_load_async_to_lds_b128 %0, %1, off"
               :: "v"(lds_dst), "v"(ga) : "memory");
  asm volatile("global_load_async_to_lds_b128 %0, %1, off offset:16"
               :: "v"(lds_dst), "v"(ga) : "memory");
  asm volatile("global_load_async_to_lds_b128 %0, %1, off offset:1024"
               :: "v"(lds_dst), "v"(ga) : "memory");
  asm volatile("global_load_async_to_lds_b128 %0, %1, off offset:1040"
               :: "v"(lds_dst), "v"(ga) : "memory");
}
template <int N>
__device__ __forceinline__ void async_wait_le() {
#if __has_builtin(__builtin_amdgcn_s_wait_asynccnt)
  __builtin_amdgcn_s_wait_asynccnt(N);
#else
  if (N == 4) asm volatile("s_wait_asynccnt 0x4" ::: "memory");
  else        asm volatile("s_wait_asynccnt 0x0" ::: "memory");
#endif
}

// ---------------------------------------------------------------- utility
__global__ void k_zero(float* p, long long n) {
  long long i = (long long)blockIdx.x * blockDim.x + threadIdx.x;
  if (i < n) p[i] = 0.f;
}

__global__ void k_init_h(const float* __restrict__ x, float* __restrict__ h) {
  int n = blockIdx.x * blockDim.x + threadIdx.x;
  if (n >= NNODEc) return;
  for (int j = 0; j < 5; ++j) h[n * 32 + j] = x[n * 15 + j];
}

// ------------------------------------------------ NNConv: pack w2 (+b2) into
// f16 WMMA-B-fragment layout: pack[((c*2+t)*32+lane)*16 + j]
__global__ void k_pack_w2(const float* __restrict__ w2, const float* __restrict__ b2,
                          h16* __restrict__ pack, int m_in) {
  int nmain = m_in * 4;                // K' = i*128 + k, chunks of 32
  int nch = nmain + 1;                 // + one bias chunk
  int gid = blockIdx.x * blockDim.x + threadIdx.x;
  if (gid >= nch * 2 * 32) return;
  int lane = gid & 31, t = (gid >> 5) & 1, c = gid >> 6;
  int n = lane & 15, hi = lane >> 4;
  int ldw = m_in * 32;
  for (int j = 0; j < 16; ++j) {
    float v;
    if (c < nmain) {
      int kg = (c & 3) * 32 + hi * 16 + j;          // k in [0,128)
      int col = (c >> 2) * 32 + t * 16 + n;         // i*32 + o
      v = w2[(long long)kg * ldw + col];
    } else {
      int jb = hi * 16 + j;                          // i index
      v = (jb < m_in) ? b2[jb * 32 + t * 16 + n] : 0.f;
    }
    pack[((long long)(c * 2 + t) * 32 + lane) * 16 + j] = (h16)v;
  }
}

// ---------- generic conv1x1 weight pack: Wp[((kc*8+t)*32+lane)*16+j], Cout=128
__global__ void k_pack_w(const float* __restrict__ W, h16* __restrict__ Wp, int Cin) {
  int kch = (Cin + 31) >> 5;
  int gid = blockIdx.x * blockDim.x + threadIdx.x;
  if (gid >= kch * 8 * 32) return;
  int lane = gid & 31, t = (gid >> 5) & 7, kc = gid >> 8;
  int n = lane & 15, hi = lane >> 4;
  for (int j = 0; j < 16; ++j) {
    int c = kc * 32 + hi * 16 + j;
    float v = (c < Cin) ? W[(long long)(t * 16 + n) * Cin + c] : 0.f;
    Wp[((long long)(kc * 8 + t) * 32 + lane) * 16 + j] = (h16)v;
  }
}

// ------------------------------------------------ fused NNConv message+scatter
// one wave per 16-edge tile; rank-1 A tiles built in registers, WMMA GEMM over
// K' = M_IN*128 (+ bias chunk). B-fragments double-buffered in LDS via
// GLOBAL_LOAD_ASYNC_TO_LDS (prefetch chunk c+1 while WMMAs run on chunk c).
template <int M_IN>
__global__ __launch_bounds__(128)
void k_nnconv(const float* __restrict__ hin, const float* __restrict__ ea,
              const float* __restrict__ w1, const float* __restrict__ b1,
              const h16* __restrict__ pack, const int* __restrict__ ei,
              float* __restrict__ agg) {
  __shared__ alignas(16) float hid_s[4][16][128];
  __shared__ alignas(16) float hsv[4][16][32];
  __shared__ alignas(32) h16 pbuf[4][2][1024];     // per-wave double buffer
  int w = threadIdx.x >> 5, lane = threadIdx.x & 31;
  int tile = blockIdx.x * 4 + w;                 // EDG/16 = 20480 tiles exactly
  int m = lane & 15, hi = lane >> 4;
  int e = tile * 16 + m;
  int src = ei[e];
  // gather h_src tile (vectorized; each lane copies its 16-column half)
  {
    const float4* hr = (const float4*)(hin + (long long)src * 32);
    float4* dsty = (float4*)&hsv[w][m][hi * 16];
#pragma unroll
    for (int q = 0; q < 4; ++q) dsty[q] = hr[hi * 4 + q];
  }
  // hid tile: hid[m][k] = relu(b1[k] + sum_j ea[e,j]*w1[j,k]); lane does 64 k's
  float a0 = ea[e * 5 + 0], a1 = ea[e * 5 + 1], a2 = ea[e * 5 + 2],
        a3 = ea[e * 5 + 3], a4 = ea[e * 5 + 4];
  for (int kk = 0; kk < 64; ++kk) {
    int k = hi * 64 + kk;
    float s = b1[k] + a0 * w1[k] + a1 * w1[128 + k] + a2 * w1[256 + k] +
              a3 * w1[384 + k] + a4 * w1[512 + k];
    hid_s[w][m][k] = fmaxf(s, 0.f);
  }
  __syncthreads();
  constexpr int NMAIN = M_IN * 4;
  constexpr int NCH = NMAIN + 1;
  unsigned mydst = (unsigned)(size_t)&pbuf[w][0][0] + (lane << 5);
  // prologue: async-prefetch chunk 0
  async_cp32B_x2(mydst, pack + (long long)lane * 16);
  v8f acc0 = {0.f, 0.f, 0.f, 0.f, 0.f, 0.f, 0.f, 0.f};
  v8f acc1 = acc0;
  for (int c = 0; c < NCH; ++c) {
    if (c + 1 < NCH) {
      async_cp32B_x2(mydst + (((c + 1) & 1) << 11),
                     pack + (long long)(c + 1) * 1024 + lane * 16);
      async_wait_le<4>();                 // 4 oldest (= chunk c) retired
    } else {
      async_wait_le<0>();
    }
    v16h a;
    if (c < NMAIN) {
      float s = hsv[w][m][c >> 2];
      const float* hp = &hid_s[w][m][(c & 3) * 32];
#pragma unroll
      for (int r = 0; r < 8; ++r) {      // k-pairs are contiguous -> float2 LDS
        float2 pr = *(const float2*)&hp[kAidx(2 * r, hi)];
        a[2 * r]     = (h16)(s * pr.x);
        a[2 * r + 1] = (h16)(s * pr.y);
      }
    } else {  // bias chunk: A[m][i] = hsv[m][i] for i<M_IN else 0 (branchless)
#pragma unroll
      for (int j = 0; j < 16; ++j) {
        int jb = kAidx(j, hi);
        float v = hsv[w][m][jb & 31];
        v = (jb < M_IN) ? v : 0.f;
        a[j] = (h16)v;
      }
    }
    const h16* bp = &pbuf[w][c & 1][lane * 16];
    v16h b0 = *(const v16h*)(bp);
    v16h b1f = *(const v16h*)(bp + 512);
    acc0 = wmma16(a, b0, acc0);
    acc1 = wmma16(a, b1f, acc1);
  }
  int n = lane & 15;
#pragma unroll
  for (int r = 0; r < 8; ++r) {
    int row = r + 8 * hi;
    int dst = ei[EDGc + tile * 16 + row];
    atomicAdd(&agg[(long long)dst * 32 + n], acc0[r]);
    atomicAdd(&agg[(long long)dst * 32 + 16 + n], acc1[r]);
  }
}

__global__ void k_finish(const float* __restrict__ hin, const float* __restrict__ agg,
                         const float* __restrict__ root, const float* __restrict__ bias,
                         float* __restrict__ hout, int m_in) {
  int gid = blockIdx.x * blockDim.x + threadIdx.x;
  if (gid >= NNODEc * 32) return;
  int node = gid >> 5, o = gid & 31;
  float s = agg[gid] + bias[o];
  for (int i = 0; i < m_in; ++i) s += hin[node * 32 + i] * root[i * 32 + o];
  hout[gid] = (s > 0.f) ? s : (expf(s) - 1.f);   // ELU
}

// ------------------------------------------------ fc1 over concat(h32, x[:,5:])
__global__ void k_fc1(const float* __restrict__ h, const float* __restrict__ x,
                      const float* __restrict__ w, const float* __restrict__ b,
                      float* __restrict__ out16) {
  int gid = blockIdx.x * blockDim.x + threadIdx.x;
  if (gid >= NNODEc * 16) return;
  int node = gid >> 4, o = gid & 15;
  float s = b[o];
  for (int c = 0; c < 32; ++c) s += h[node * 32 + c] * w[c * 16 + o];
  for (int c = 0; c < 10; ++c) s += x[node * 15 + 5 + c] * w[(32 + c) * 16 + o];
  out16[gid] = s;
}

__global__ void k_mean(const float* __restrict__ hn, float* __restrict__ dn) {
  int gid = blockIdx.x * blockDim.x + threadIdx.x;
  if (gid >= NSUBc * 16) return;
  int sub = gid >> 4, o = gid & 15;
  float s = 0.f;
  for (int k = 0; k < SSp; ++k) s += hn[(sub * SSp + k) * 16 + o];
  dn[gid] = s * 0.1f;
}

// ------------------------------------------------ build z (B,22,64,64); layout
// z[b,c,j,i], plane-flat p = j*64+i (already the transposed (0,3,2,1) layout)
__global__ void k_fill_z(const float* __restrict__ opos, const float* __restrict__ dn,
                         float* __restrict__ z) {
  int gid = blockIdx.x * blockDim.x + threadIdx.x;
  if (gid >= BB * 4096) return;
  int b = gid >> 12, rem = gid & 4095, j = rem >> 6, i = rem & 63;
  const float* pi = opos + (long long)(b * 64 + i) * 3;
  const float* pj = opos + (long long)(b * 64 + j) * 3;
  float dx = pi[0] - pj[0], dy = pi[1] - pj[1], dz = pi[2] - pj[2];
  float sq = dx * dx + dy * dy + dz * dz;
  z[((long long)(b * 22 + 5) << 12) + rem] = (sq > 0.f) ? sqrtf(sq) : 0.f;
  for (int k = 0; k < 16; ++k)
    z[((long long)(b * 22 + 6 + k) << 12) + rem] =
        (i == j) ? dn[(b * 64 + i) * 16 + k] : 0.f;
}

__global__ void k_scatter(const float* __restrict__ oea, const int* __restrict__ oei,
                          const int* __restrict__ s2g, float* __restrict__ z) {
  int e = blockIdx.x * blockDim.x + threadIdx.x;
  if (e >= EORIGc) return;
  int sn = oei[e], tn = oei[EORIGc + e];
  int g = s2g[sn];
  int ls = sn - g * 64, lt = tn - g * 64;
  long long base = ((long long)(g * 22) << 12) + lt * 64 + ls;  // z[g,c,lt,ls]
  for (int c = 0; c < 4; ++c)
    atomicAdd(&z[base + ((long long)c << 12)], oea[e * 5 + c]);
  atomicAdd(&z[base + ((long long)4 << 12)], 1.0f);
}

// ------------------------------------------------ conv1x1 as WMMA GEMM
// block stages a 64-row x Cin activation tile in LDS (coalesced, f32->f16,
// zero-padded tail); 4 waves compute 16 rows x 128 outs each. Cout = 128.
template <int C1, int C2>
__global__ __launch_bounds__(128)
void k_conv1x1(const float* __restrict__ X1, const float* __restrict__ X2,
               const h16* __restrict__ Wp, const float* __restrict__ bias,
               float* __restrict__ out, int relu) {
  constexpr int Cin = C1 + C2;
  constexpr int KCH = (Cin + 31) >> 5;
  constexpr int CP = KCH * 32 + 8;                 // padded LDS row stride
  __shared__ alignas(32) h16 As[64 * CP];
  int rowbase = blockIdx.x * 64;                   // 4096 blocks, same b per blk
  int b = rowbase >> 12, p0 = rowbase & 4095;
  const float* x1b = X1 + (((long long)b * C1) << 12) + p0;
  const float* x2b = X2 + (((long long)b * C2) << 12) + p0;
  for (int idx = threadIdx.x; idx < 64 * KCH * 32; idx += 128) {
    int c = idx >> 6, r = idx & 63;                // coalesced 64-float runs
    int c2 = c - C1;
    int cx = (c < C1) ? c : ((c2 < C2) ? c2 : 0);
    const float* srcp = (c < C1) ? x1b : x2b;
    float v = srcp[((long long)cx << 12) + r];
    if (Cin % 32) v = (c < Cin) ? v : 0.f;         // tail mask (compile-time)
    As[r * CP + c] = (h16)v;
  }
  __syncthreads();
  int wv = threadIdx.x >> 5, lane = threadIdx.x & 31;
  int m = lane & 15, hi = lane >> 4, n = m;
  int roff = wv * 16;
  v8f z8 = {0.f, 0.f, 0.f, 0.f, 0.f, 0.f, 0.f, 0.f};
  v8f acc[8];
#pragma unroll
  for (int t = 0; t < 8; ++t) acc[t] = z8;
#pragma unroll
  for (int kc = 0; kc < KCH; ++kc) {
    const h16* ar = &As[(roff + m) * CP + kc * 32];
    h16v8 lo = *(const h16v8*)(ar + 16 * hi);
    h16v8 up = *(const h16v8*)(ar + 8 + 16 * hi);
    v16h a;
#pragma unroll
    for (int j = 0; j < 8; ++j) { a[j] = lo[j]; a[8 + j] = up[j]; }
    const h16* wp = Wp + (long long)kc * 4096 + (long long)lane * 16;
#pragma unroll
    for (int t = 0; t < 8; ++t) {
      v16h bf = *(const v16h*)(wp + t * 512);
      acc[t] = wmma16(a, bf, acc[t]);
    }
  }
#pragma unroll
  for (int t = 0; t < 8; ++t) {
#pragma unroll
    for (int r = 0; r < 8; ++r) {
      int row = roff + r + 8 * hi, o = t * 16 + n;
      float v = acc[t][r] + bias[o];
      if (relu) v = fmaxf(v, 0.f);
      out[(((long long)b * 128 + o) << 12) + p0 + row] = v;
    }
  }
}

// ------------------------------------------------ per-(b,c) 64x64x64 matmul
#define BS_LD 80   /* padded LDS stride (halves); 160B keeps 32B-aligned cols */
__global__ __launch_bounds__(128)
void k_bmm(const float* __restrict__ m1, const float* __restrict__ m2,
           float* __restrict__ out) {
  __shared__ alignas(32) h16 As[4096];
  __shared__ alignas(32) h16 Bs[64 * BS_LD];      // transposed: Bs[col*LD + k]
  long long off = (long long)blockIdx.x << 12;
  for (int i = 0; i < 32; ++i) {
    int idx = threadIdx.x + (i << 7);
    int r = idx >> 6, ccol = idx & 63;
    As[idx] = (h16)m1[off + idx];
    Bs[ccol * BS_LD + r] = (h16)m2[off + idx];
  }
  __syncthreads();
  int wv = threadIdx.x >> 5, lane = threadIdx.x & 31;
  int m = lane & 15, hi = lane >> 4, n = m;
  int rowb = wv * 16;
  const h16* ar = &As[(rowb + m) * 64];
  v16h a0, a1;
  {
    h16v8 l0 = *(const h16v8*)(ar + 16 * hi);
    h16v8 u0 = *(const h16v8*)(ar + 8 + 16 * hi);
    h16v8 l1 = *(const h16v8*)(ar + 32 + 16 * hi);
    h16v8 u1 = *(const h16v8*)(ar + 40 + 16 * hi);
#pragma unroll
    for (int j = 0; j < 8; ++j) {
      a0[j] = l0[j]; a0[8 + j] = u0[j];
      a1[j] = l1[j]; a1[8 + j] = u1[j];
    }
  }
#pragma unroll
  for (int ct = 0; ct < 4; ++ct) {
    const h16* bc = &Bs[(ct * 16 + n) * BS_LD + hi * 16];
    v16h b0 = *(const v16h*)(bc);        // k = hi*16 + j      (kc=0)
    v16h b1 = *(const v16h*)(bc + 32);   // k = 32 + hi*16 + j (kc=1)
    v8f acc = {0.f, 0.f, 0.f, 0.f, 0.f, 0.f, 0.f, 0.f};
    acc = wmma16(a0, b0, acc);
    acc = wmma16(a1, b1, acc);
#pragma unroll
    for (int r = 0; r < 8; ++r) {
      int row = rowb + r + 8 * hi;
      out[off + row * 64 + ct * 16 + n] = acc[r];
    }
  }
}

// ------------------------------------------------ tail reductions + MLP
__global__ void k_pass1(const float* __restrict__ z, float* __restrict__ md,
                        float* __restrict__ mo, float* __restrict__ mn) {
  int gw = blockIdx.x * 8 + (threadIdx.x >> 5);   // one wave per (b,c), 8192
  int lane = threadIdx.x & 31;
  long long plane = (long long)gw << 12;
  float dmax = -3.4e38f, omax = -3.4e38f, nmax = -3.4e38f;
  for (int p = lane; p < 4096; p += 32) {
    float v = z[plane + p];
    if ((p & 63) == (p >> 6)) dmax = fmaxf(dmax, v);
    else                      omax = fmaxf(omax, v);
    nmax = fmaxf(nmax, -v);
  }
  for (int s = 16; s > 0; s >>= 1) {
    dmax = fmaxf(dmax, __shfl_xor(dmax, s, 32));
    omax = fmaxf(omax, __shfl_xor(omax, s, 32));
    nmax = fmaxf(nmax, __shfl_xor(nmax, s, 32));
  }
  if (lane == 0) { md[gw] = dmax; mo[gw] = omax; mn[gw] = nmax; }
}

__global__ void k_pass2(const float* __restrict__ md, const float* __restrict__ mo,
                        const float* __restrict__ mn, float* __restrict__ zz) {
  __shared__ float r1[256], r2[256];
  int tid = threadIdx.x;
  float a = -3.4e38f, b = -3.4e38f;
  for (int i = tid; i < 8192; i += 256) {
    a = fmaxf(a, md[i]);
    b = fmaxf(b, mn[i]);
  }
  r1[tid] = a; r2[tid] = b;
  __syncthreads();
  for (int s = 128; s > 0; s >>= 1) {
    if (tid < s) { r1[tid] = fmaxf(r1[tid], r1[tid + s]);
                   r2[tid] = fmaxf(r2[tid], r2[tid + s]); }
    __syncthreads();
  }
  float val = fabsf(r1[0] + r2[0]);
  for (int i = tid; i < 8192; i += 256) {
    int bi = i >> 7, c = i & 127;
    zz[bi * 256 + c] = md[i];
    zz[bi * 256 + 128 + c] = fmaxf(mo[i], md[i] - val);
  }
}

__global__ void k_pass3(const float* __restrict__ zz, const float* __restrict__ w1,
                        const float* __restrict__ b1, const float* __restrict__ w2,
                        const float* __restrict__ b2, float* __restrict__ out) {
  __shared__ float h1[64];
  int b = blockIdx.x, o = threadIdx.x;
  float s = b1[o];
  for (int c = 0; c < 256; ++c) s += zz[b * 256 + c] * w1[c * 64 + o];
  h1[o] = fmaxf(s, 0.f);
  __syncthreads();
  if (o == 0) {
    float r = b2[0];
    for (int k = 0; k < 64; ++k) r += h1[k] * w2[k];
    out[b] = r;
  }
}

// ================================================================= launcher
extern "C" void kernel_launch(void* const* d_in, const int* in_sizes, int n_in,
                              void* d_out, int out_size, void* d_ws, size_t ws_size,
                              hipStream_t stream) {
  // input indices (jax pytree flatten: dict keys sorted, tuples in order)
  const float* x    = (const float*)d_in[0];
  const float* ea   = (const float*)d_in[1];
  const float* oea  = (const float*)d_in[2];
  const float* opos = (const float*)d_in[3];
  // convs[l]: b1,b2,bias,root,w1,w2 at 4+6l
  const float* fc1w = (const float*)d_in[22];
  const float* fc1b = (const float*)d_in[23];
  const float* fp1w = (const float*)d_in[24];
  const float* fp1b = (const float*)d_in[25];
  const float* fp2w = (const float*)d_in[26];
  const float* fp2b = (const float*)d_in[27];
  const int* ei  = (const int*)d_in[48];
  const int* oei = (const int*)d_in[50];
  const int* s2g = (const int*)d_in[51];
  float* out = (float*)d_out;

  // workspace carve-up
  char* ws = (char*)d_ws;
  size_t off = 0;
  auto take = [&](size_t bytes) { char* p = ws + off; off += (bytes + 255) & ~(size_t)255; return p; };
  float* hA    = (float*)take((size_t)NNODEc * 32 * 4);
  float* hB    = (float*)take((size_t)NNODEc * 32 * 4);
  float* agg   = (float*)take((size_t)NNODEc * 32 * 4);
  float* hn16  = (float*)take((size_t)NNODEc * 16 * 4);
  float* dn    = (float*)take((size_t)NSUBc * 16 * 4);
  h16*   pack  = (h16*)  take((size_t)129 * 1024 * 2);
  h16*   Wp    = (h16*)  take((size_t)8 * 4096 * 2);   // max Cin=256 -> 64KB
  float* z22   = (float*)take((size_t)BB * 22 * 4096 * 4);
  float* Pb    = (float*)take((size_t)BB * 128 * 4096 * 4);
  float* Qb    = (float*)take((size_t)BB * 128 * 4096 * 4);
  float* Rb    = (float*)take((size_t)BB * 128 * 4096 * 4);
  float* Tb    = (float*)take((size_t)BB * 128 * 4096 * 4);
  float* md    = (float*)take(8192 * 4);
  float* mo    = (float*)take(8192 * 4);
  float* mn    = (float*)take(8192 * 4);
  float* zz    = (float*)take(64 * 256 * 4);
  (void)in_sizes; (void)n_in; (void)out_size; (void)ws_size;

  auto packw = [&](int idx, int Cin) {
    k_pack_w<<<(((Cin + 31) >> 5) * 256 + 255) / 256, 256, 0, stream>>>(
        (const float*)d_in[idx], Wp, Cin);
  };

  // ---- 3 NNConv + ELU layers
  k_init_h<<<160, 256, 0, stream>>>(x, hA);
  const float* hin = hA;
  float* hout = hB;
  for (int l = 0; l < 3; ++l) {
    int m_in = (l == 0) ? 5 : 32;
    int base = 4 + l * 6;
    const float* cb1   = (const float*)d_in[base + 0];
    const float* cb2   = (const float*)d_in[base + 1];
    const float* cbias = (const float*)d_in[base + 2];
    const float* croot = (const float*)d_in[base + 3];
    const float* cw1   = (const float*)d_in[base + 4];
    const float* cw2   = (const float*)d_in[base + 5];
    int nch = m_in * 4 + 1;
    k_zero<<<5120, 256, 0, stream>>>(agg, (long long)NNODEc * 32);
    k_pack_w2<<<(nch * 64 + 255) / 256, 256, 0, stream>>>(cw2, cb2, pack, m_in);
    if (m_in == 5)
      k_nnconv<5><<<5120, 128, 0, stream>>>(hin, ea, cw1, cb1, pack, ei, agg);
    else
      k_nnconv<32><<<5120, 128, 0, stream>>>(hin, ea, cw1, cb1, pack, ei, agg);
    k_finish<<<5120, 256, 0, stream>>>(hin, agg, croot, cbias, hout, m_in);
    const float* t = hin; hin = hout; hout = (float*)t;
  }
  // ---- fc1 + subgraph mean pooling
  k_fc1<<<2560, 256, 0, stream>>>(hin, x, fc1w, fc1b, hn16);
  k_mean<<<256, 256, 0, stream>>>(hn16, dn);
  // ---- build z (B,22,64,64)
  k_zero<<<(BB * 22 * 4096 + 255) / 256, 256, 0, stream>>>(z22, (long long)BB * 22 * 4096);
  k_fill_z<<<1024, 256, 0, stream>>>(opos, dn, z22);
  k_scatter<<<256, 256, 0, stream>>>(oea, oei, s2g, z22);
  // ---- regular block 1 (in: z22[22ch])
  packw(28, 22);
  k_conv1x1<22, 0><<<4096, 128, 0, stream>>>(z22, z22, Wp, (const float*)d_in[29], Pb, 1);
  packw(30, 128);
  k_conv1x1<128, 0><<<4096, 128, 0, stream>>>(Pb, Pb, Wp, (const float*)d_in[31], Qb, 1);
  packw(32, 22);
  k_conv1x1<22, 0><<<4096, 128, 0, stream>>>(z22, z22, Wp, (const float*)d_in[33], Rb, 1);
  packw(34, 128);
  k_conv1x1<128, 0><<<4096, 128, 0, stream>>>(Rb, Rb, Wp, (const float*)d_in[35], Pb, 1);
  k_bmm<<<8192, 128, 0, stream>>>(Qb, Pb, Rb);                     // mult -> R
  packw(36, 150);
  k_conv1x1<22, 128><<<4096, 128, 0, stream>>>(z22, Rb, Wp, (const float*)d_in[37], Tb, 0);
  // ---- regular block 2 (in: Tb[128ch])
  packw(38, 128);
  k_conv1x1<128, 0><<<4096, 128, 0, stream>>>(Tb, Tb, Wp, (const float*)d_in[39], Pb, 1);
  packw(40, 128);
  k_conv1x1<128, 0><<<4096, 128, 0, stream>>>(Pb, Pb, Wp, (const float*)d_in[41], Qb, 1);
  packw(42, 128);
  k_conv1x1<128, 0><<<4096, 128, 0, stream>>>(Tb, Tb, Wp, (const float*)d_in[43], Pb, 1);
  packw(44, 128);
  k_conv1x1<128, 0><<<4096, 128, 0, stream>>>(Pb, Pb, Wp, (const float*)d_in[45], Rb, 1);
  k_bmm<<<8192, 128, 0, stream>>>(Qb, Rb, Pb);                     // mult -> P
  packw(46, 256);
  k_conv1x1<128, 128><<<4096, 128, 0, stream>>>(Tb, Pb, Wp, (const float*)d_in[47], Qb, 0);
  // ---- tail: diag/off-diag maxima + final MLP
  k_pass1<<<1024, 256, 0, stream>>>(Qb, md, mo, mn);
  k_pass2<<<1, 256, 0, stream>>>(md, mo, mn, zz);
  k_pass3<<<64, 64, 0, stream>>>(zz, fp1w, fp1b, fp2w, fp2b, out);
}
